// GraphAttentionLayer_7129645712022
// MI455X (gfx1250) — matrix operand order repaired and checked by reading, not compiled, not run on hardware
//
#include <hip/hip_runtime.h>
#include <hip/hip_bf16.h>

// ---------------------------------------------------------------------------
// GAT layer, MI455X (gfx1250, wave32).
//   B=8, N=2048, F=128.
//   Pass A: Wh = h @ W           via v_wmma_f32_16x16x4_f32 (2 interleaved
//                                 chains per wave for ILP; fp32 kept)
//   Pass B: s1 = Wh@a1, s2=Wh@a2 (wave-shuffle dot products)
//   Pass C: per-batch max(s2)    -> safe softmax bound m_i = lrelu(s1_i+max s2)
//   Pass D: single pass over adj (134 MB @ 23.3 TB/s ~= 5.8us floor):
//           p_ij = adj ? exp(lrelu(s1_i+s2_j) - m_i) : 0   (always <= 1)
//           h' = (P @ Wh) / rowsum(P)   via v_wmma_f32_16x16x32_bf16
//           then ELU.  m_i is a fixed bound -> no online-softmax rescale,
//           adj read exactly once (non-temporal + prefetch).
// ---------------------------------------------------------------------------

#define ALPHA 0.2f
#define Bn    8
#define Nn    2048
#define Fn    128

typedef __attribute__((ext_vector_type(2)))  float  v2f;
typedef __attribute__((ext_vector_type(8)))  float  v8f;
typedef __attribute__((ext_vector_type(4)))  float  v4f;
typedef __attribute__((ext_vector_type(4)))  int    v4i;
typedef __attribute__((ext_vector_type(16))) __bf16 v16bf;
typedef __attribute__((ext_vector_type(8)))  __bf16 v8bf;

static __device__ __forceinline__ float lrelu(float x) { return x > 0.f ? x : ALPHA * x; }

// ---------------- Kernel 1: Wh = h @ W (fp32 WMMA), also store WhT in bf16 --
// grid 1024 blocks x 128 thr; block owns 16 flat rows; wave runs 2 col-tiles
// as two independent interleaved WMMA accumulate chains (shared A operand).
__global__ __launch_bounds__(128) void k_proj(const float* __restrict__ h,
                                              const float* __restrict__ W,
                                              float* __restrict__ Wh,
                                              __bf16* __restrict__ WhT) {
    const int row_base = blockIdx.x * 16;          // flat row in [0,16384)
    const int wave = threadIdx.x >> 5;
    const int lane = threadIdx.x & 31;
    const int mn = lane & 15;                      // A-row / B-col / C-col
    const int g  = lane >> 4;
    const int bidx = row_base >> 11;               // batch
    const int rloc = row_base & 2047;              // row within batch

    const float* hrow = h + (size_t)(row_base + mn) * Fn;
    const int colA = wave * 16;        // col tiles 0..3
    const int colB = wave * 16 + 64;   // col tiles 4..7

    v8f acc0 = {}, acc1 = {};
#pragma unroll
    for (int k0 = 0; k0 < Fn; k0 += 4) {
        // A 16x4 f32: lane(m,g) holds K = 2g, 2g+1  (shared by both tiles)
        float2 av = *(const float2*)(hrow + k0 + 2 * g);
        v2f A; A.x = av.x; A.y = av.y;
        // B 4x16 f32: lane(n,g) holds K = 2g, 2g+1
        const float* wk0 = W + (size_t)(k0 + 2 * g) * Fn;
        const float* wk1 = W + (size_t)(k0 + 2 * g + 1) * Fn;
        v2f B0; B0.x = wk0[colA + mn]; B0.y = wk1[colA + mn];
        v2f B1; B1.x = wk0[colB + mn]; B1.y = wk1[colB + mn];
        acc0 = __builtin_amdgcn_wmma_f32_16x16x4_f32(false, A, false, B0,
                                                     (short)0, acc0, false, false);
        acc1 = __builtin_amdgcn_wmma_f32_16x16x4_f32(false, A, false, B1,
                                                     (short)0, acc1, false, false);
    }

    // C 16x16 f32: element r -> row (r+8g), col (col0+mn)
    v8bf tv0, tv1;
#pragma unroll
    for (int r = 0; r < 8; ++r) {
        Wh[(size_t)(row_base + r + 8 * g) * Fn + colA + mn] = acc0[r];
        Wh[(size_t)(row_base + r + 8 * g) * Fn + colB + mn] = acc1[r];
        tv0[r] = (__bf16)acc0[r];
        tv1[r] = (__bf16)acc1[r];
    }
    // WhT[b][col][row] : 8 consecutive rows -> one 16B store per tile
    __bf16* tb = WhT + (size_t)bidx * Fn * Nn + rloc + 8 * g;
    *(v8bf*)(tb + (size_t)(colA + mn) * Nn) = tv0;
    *(v8bf*)(tb + (size_t)(colB + mn) * Nn) = tv1;
}

// ---------------- Kernel 2: s1 = Wh@a1, s2 = Wh@a2 --------------------------
// grid 2048 x 256 thr; one wave per row.
__global__ __launch_bounds__(256) void k_scores(const float* __restrict__ Wh,
                                                const float* __restrict__ a,
                                                float* __restrict__ s1,
                                                float* __restrict__ s2) {
    const int wave = threadIdx.x >> 5;
    const int lane = threadIdx.x & 31;
    const int row  = blockIdx.x * 8 + wave;
    const float* wr = Wh + (size_t)row * Fn;
    float acc1 = 0.f, acc2 = 0.f;
#pragma unroll
    for (int i = 0; i < 4; ++i) {
        int k = i * 32 + lane;
        float w = wr[k];
        acc1 += w * a[k];
        acc2 += w * a[Fn + k];
    }
#pragma unroll
    for (int off = 16; off > 0; off >>= 1) {
        acc1 += __shfl_xor(acc1, off);
        acc2 += __shfl_xor(acc2, off);
    }
    if (lane == 0) { s1[row] = acc1; s2[row] = acc2; }
}

// ---------------- Kernel 3: per-batch max of s2 -----------------------------
__global__ __launch_bounds__(256) void k_s2max(const float* __restrict__ s2,
                                               float* __restrict__ s2max) {
    __shared__ float red[8];
    const int b = blockIdx.x;
    const float* sb = s2 + (size_t)b * Nn;
    float mx = -3.0e38f;
    for (int j = threadIdx.x; j < Nn; j += 256) mx = fmaxf(mx, sb[j]);
#pragma unroll
    for (int off = 16; off > 0; off >>= 1) mx = fmaxf(mx, __shfl_xor(mx, off));
    if ((threadIdx.x & 31) == 0) red[threadIdx.x >> 5] = mx;
    __syncthreads();
    if (threadIdx.x == 0) {
        for (int i = 1; i < 8; ++i) mx = fmaxf(mx, red[i]);
        s2max[b] = mx;
    }
}

// ---------------- Kernel 4: fused mask/exp + P@Wh (bf16 WMMA) + ELU ---------
// grid 1024 x 128 thr (4 waves); block = 16 rows of one batch; waves split j.
// adj is streamed non-temporally (read once); WhT stays L2-resident.
__global__ __launch_bounds__(128) void k_attn(const int* __restrict__ adj,
                                              const __bf16* __restrict__ WhT,
                                              const float* __restrict__ s1,
                                              const float* __restrict__ s2,
                                              const float* __restrict__ s2max,
                                              float* __restrict__ out) {
    __shared__ float s2sh[Nn];
    __shared__ float accsh[4 * 16 * Fn];   // per-wave partial 16x128 tiles
    __shared__ float rowsumsh[4 * 2 * 16]; // [wave][g][m]

    const int b    = blockIdx.x >> 7;        // /128
    const int row0 = (blockIdx.x & 127) * 16;
    const int wave = threadIdx.x >> 5;
    const int lane = threadIdx.x & 31;
    const int mn = lane & 15;
    const int g  = lane >> 4;

    // stage s2[b][*] into LDS (coalesced)
    const float* s2b = s2 + (size_t)b * Nn;
#pragma unroll
    for (int i = 0; i < 4; ++i) {
        int idx = (i * 128 + threadIdx.x) * 4;
        *(v4f*)&s2sh[idx] = *(const v4f*)&s2b[idx];
    }
    __syncthreads();

    const float s1v  = s1[(size_t)b * Nn + row0 + mn];
    const float mrow = lrelu(s1v + s2max[b]);

    const int* adjRow = adj + ((size_t)b * Nn + row0 + mn) * Nn;
    const __bf16* whTb = WhT + (size_t)b * Fn * Nn;

    v8f acc[8];
#pragma unroll
    for (int c = 0; c < 8; ++c) acc[c] = (v8f){};
    float rsum = 0.f;

    for (int jt = wave; jt < Nn / 32; jt += 4) {
        const int j0 = jt * 32;
        // prefetch this wave's next adj tile (j0 advances by 128 ints = 512B)
        if (jt + 4 < Nn / 32) __builtin_prefetch(adjRow + j0 + 128, 0, 0);

        // A-layout adj fetch (non-temporal, single-use stream):
        // K = {8g+0..7} u {16+8g+0..7}
        v4i q0 = __builtin_nontemporal_load((const v4i*)(adjRow + j0 + 8 * g));
        v4i q1 = __builtin_nontemporal_load((const v4i*)(adjRow + j0 + 8 * g + 4));
        v4i q2 = __builtin_nontemporal_load((const v4i*)(adjRow + j0 + 16 + 8 * g));
        v4i q3 = __builtin_nontemporal_load((const v4i*)(adjRow + j0 + 20 + 8 * g));
        v4f u0 = *(const v4f*)&s2sh[j0 + 8 * g];
        v4f u1 = *(const v4f*)&s2sh[j0 + 8 * g + 4];
        v4f u2 = *(const v4f*)&s2sh[j0 + 16 + 8 * g];
        v4f u3 = *(const v4f*)&s2sh[j0 + 20 + 8 * g];

        int   aj[16] = {q0.x,q0.y,q0.z,q0.w, q1.x,q1.y,q1.z,q1.w,
                        q2.x,q2.y,q2.z,q2.w, q3.x,q3.y,q3.z,q3.w};
        float sv[16] = {u0.x,u0.y,u0.z,u0.w, u1.x,u1.y,u1.z,u1.w,
                        u2.x,u2.y,u2.z,u2.w, u3.x,u3.y,u3.z,u3.w};

        v16bf A;
#pragma unroll
        for (int t = 0; t < 16; ++t) {
            float e = lrelu(s1v + sv[t]);
            float p = aj[t] > 0 ? __expf(e - mrow) : 0.f;  // always <= 1
            rsum += p;
            A[t] = (__bf16)p;
        }
        // B 32x16 bf16: lane(n,g) needs K = {8g+0..7} u {16+8g+0..7} from WhT[col][j0+K]
#pragma unroll
        for (int c = 0; c < 8; ++c) {
            const __bf16* bp = whTb + (size_t)(c * 16 + mn) * Nn + j0 + 8 * g;
            const v8bf b0 = *(const v8bf*)(bp);
            const v8bf b1 = *(const v8bf*)(bp + 16);
            const v16bf Bv = __builtin_shufflevector(b0, b1, 0,1,2,3,4,5,6,7,8,9,10,11,12,13,14,15);
            acc[c] = __builtin_amdgcn_wmma_f32_16x16x32_bf16(false, A, false, Bv,
                                                             (short)0, acc[c],
                                                             false, false);
        }
    }

    // deposit per-wave partials: C element r -> row (r+8g), col (c*16+mn)
    float* ap = accsh + wave * (16 * Fn);
#pragma unroll
    for (int c = 0; c < 8; ++c)
#pragma unroll
        for (int r = 0; r < 8; ++r)
            ap[(r + 8 * g) * Fn + c * 16 + mn] = acc[c][r];
    rowsumsh[wave * 32 + g * 16 + mn] = rsum;
    __syncthreads();

    // combine partials, normalize, ELU, store (col = threadIdx.x, coalesced)
    const int col = threadIdx.x;
    for (int rr = 0; rr < 16; ++rr) {
        float sum = accsh[0 * 2048 + rr * Fn + col] + accsh[1 * 2048 + rr * Fn + col]
                  + accsh[2 * 2048 + rr * Fn + col] + accsh[3 * 2048 + rr * Fn + col];
        float rs = 0.f;
#pragma unroll
        for (int w = 0; w < 4; ++w)
            rs += rowsumsh[w * 32 + rr] + rowsumsh[w * 32 + 16 + rr];
        float v = sum / rs;
        v = v > 0.f ? v : (__expf(v) - 1.f);
        __builtin_nontemporal_store(v, &out[((size_t)b * Nn + row0 + rr) * Fn + col]);
    }
}

// ---------------------------------------------------------------------------
extern "C" void kernel_launch(void* const* d_in, const int* in_sizes, int n_in,
                              void* d_out, int out_size, void* d_ws, size_t ws_size,
                              hipStream_t stream) {
    const float* h   = (const float*)d_in[0];
    const int*   adj = (const int*)  d_in[1];
    const float* W   = (const float*)d_in[2];
    const float* a   = (const float*)d_in[3];
    float* out = (float*)d_out;

    char* ws = (char*)d_ws;
    float*  Wh    = (float*) (ws + 0);                  // 16384*128*4  = 8 MiB
    __bf16* WhT   = (__bf16*)(ws + 8388608);            // 16384*128*2  = 4 MiB
    float*  s1    = (float*) (ws + 12582912);           // 64 KiB
    float*  s2    = (float*) (ws + 12648448);           // 64 KiB
    float*  s2max = (float*) (ws + 12713984);           // 32 B

    k_proj  <<<dim3(1024), dim3(128), 0, stream>>>(h, W, Wh, WhT);
    k_scores<<<dim3(2048), dim3(256), 0, stream>>>(Wh, a, s1, s2);
    k_s2max <<<dim3(8),    dim3(256), 0, stream>>>(s2, s2max);
    k_attn  <<<dim3(1024), dim3(128), 0, stream>>>(adj, WhT, s1, s2, s2max, out);
}
